// MultiHeadAttention_17557826306125
// MI455X (gfx1250) — compile-verified, matrix operand order checked by reading, not run on hardware
//
#include <hip/hip_runtime.h>

#define EMB   1024
#define HEADS 16
#define HDIM  64
#define NB    2
#define NT    2048
#define BT    (NB * NT)   // 4096

typedef __attribute__((ext_vector_type(16))) __bf16 v16bf;
typedef __attribute__((ext_vector_type(8)))  float  v8f;
typedef int v4i __attribute__((ext_vector_type(4)));
typedef __attribute__((address_space(1))) v4i* gptr_v4i;
typedef __attribute__((address_space(3))) v4i* lptr_v4i;

struct Frag { union { v16bf v; uint4 q[2]; }; };

__device__ inline unsigned short f2bf(float f) {
  unsigned int u = __float_as_uint(f);
  u += 0x7FFFu + ((u >> 16) & 1u);   // round-to-nearest-even
  return (unsigned short)(u >> 16);
}

__device__ inline void frag_load(Frag& f, const unsigned short* p) {
  f.q[0] = *(const uint4*)(p);
  f.q[1] = *(const uint4*)(p + 16);
}

__device__ inline v8f wmma_bf16(const Frag& a, const Frag& b, v8f c) {
  return __builtin_amdgcn_wmma_f32_16x16x32_bf16(false, a.v, false, b.v,
                                                 (short)0, c, false, false);
}

// ---- async global -> LDS copy (ASYNCcnt path)
__device__ inline void async_copy_b128(const unsigned short* g, unsigned short* l) {
#if __has_builtin(__builtin_amdgcn_global_load_async_to_lds_b128)
  __builtin_amdgcn_global_load_async_to_lds_b128(
      (gptr_v4i)(unsigned long long)g,
      (lptr_v4i)(unsigned)(unsigned long long)l, 0, 0);
#else
  unsigned loff = (unsigned)(unsigned long long)l;
  unsigned long long ga = (unsigned long long)g;
  asm volatile("global_load_async_to_lds_b128 %0, %1, off"
               :: "v"(loff), "v"(ga) : "memory");
#endif
}

__device__ inline void wait_async0() {
#if __has_builtin(__builtin_amdgcn_s_wait_asynccnt)
  __builtin_amdgcn_s_wait_asynccnt(0);
#else
  asm volatile("s_wait_asynccnt 0x0" ::: "memory");
#endif
}

// ---------------------------------------------------------------- convert
__global__ __launch_bounds__(256) void cvt_bf16(const float* __restrict__ s,
                                                unsigned short* __restrict__ d,
                                                int n) {
  int i = blockIdx.x * 256 + threadIdx.x;
  if (i < n) d[i] = f2bf(s[i]);
}

// ------------------------------------------------------- fused QKV GEMM
// grid: (BT/128, EMB/128, 3)  block: 256 (8 waves as 4m x 2n, 32x64 per wave)
// out layout: [B][H][T][D] bf16
__global__ __launch_bounds__(256) void qkv_gemm(
    const unsigned short* __restrict__ xb,
    const unsigned short* __restrict__ Wqb, const unsigned short* __restrict__ Wkb,
    const unsigned short* __restrict__ Wvb,
    const float* __restrict__ bq, const float* __restrict__ bk,
    const float* __restrict__ bv,
    unsigned short* __restrict__ Qb, unsigned short* __restrict__ Kb,
    unsigned short* __restrict__ Vb) {
  __shared__ __align__(16) unsigned short At[128 * 32];   // [m][k]
  __shared__ __align__(16) unsigned short Bt[128 * 32];   // transposed [n][k]

  const unsigned short* W = (blockIdx.z == 0) ? Wqb : (blockIdx.z == 1) ? Wkb : Wvb;
  const float* bias        = (blockIdx.z == 0) ? bq  : (blockIdx.z == 1) ? bk  : bv;
  unsigned short* dst      = (blockIdx.z == 0) ? Qb  : (blockIdx.z == 1) ? Kb  : Vb;

  const int tid = threadIdx.x;
  const int laneLo = tid & 15;
  const int hi8 = ((tid >> 4) & 1) * 8;
  const int wm = (tid >> 5) & 3;     // wave row-group (32 rows)
  const int wn = tid >> 7;           // wave col-group (64 cols)
  const int m0 = blockIdx.x * 128;
  const int n0 = blockIdx.y * 128;

  const v8f vzero = {0.f, 0.f, 0.f, 0.f, 0.f, 0.f, 0.f, 0.f};
  v8f acc[2][4];
#pragma unroll
  for (int s = 0; s < 2; ++s)
#pragma unroll
    for (int i = 0; i < 4; ++i) acc[s][i] = vzero;

  for (int k0 = 0; k0 < EMB; k0 += 32) {
    __syncthreads();
    // A tile: 128x32, async global->LDS b128 (2 chunks/thread)
#pragma unroll
    for (int i = 0; i < 2; ++i) {
      int c = tid + i * 256;               // 512 chunks of 8 bf16
      int row = c >> 2;
      int kc = (c & 3) * 8;
      async_copy_b128(xb + (size_t)(m0 + row) * EMB + k0 + kc,
                      At + row * 32 + kc);
    }
    // B tile transposed: Bt[n][k]  (2 chunks/thread, scatter b16)
#pragma unroll
    for (int i = 0; i < 2; ++i) {
      int c = tid + i * 256;               // 512 chunks
      int k = c >> 4;                      // 0..31
      int nc = (c & 15) * 8;               // 0..120
      uint4 vv = *(const uint4*)(W + (size_t)(k0 + k) * EMB + n0 + nc);
      const unsigned short* s = (const unsigned short*)&vv;
#pragma unroll
      for (int j = 0; j < 8; ++j) Bt[(nc + j) * 32 + k] = s[j];
    }
    wait_async0();
    __syncthreads();

    Frag a[2];
    frag_load(a[0], At + (wm * 32 + laneLo) * 32 + hi8);
    frag_load(a[1], At + (wm * 32 + 16 + laneLo) * 32 + hi8);
#pragma unroll
    for (int nt = 0; nt < 4; ++nt) {
      Frag b;
      frag_load(b, Bt + (wn * 64 + nt * 16 + laneLo) * 32 + hi8);
      acc[0][nt] = wmma_bf16(a[0], b, acc[0][nt]);
      acc[1][nt] = wmma_bf16(a[1], b, acc[1][nt]);
    }
  }

  // epilogue: bias + scatter to [B][H][T][D] bf16
#pragma unroll
  for (int s = 0; s < 2; ++s)
#pragma unroll
    for (int nt = 0; nt < 4; ++nt) {
      int n = n0 + wn * 64 + nt * 16 + laneLo;
      int h = n >> 6, d = n & 63;
      float bvv = bias[n];
#pragma unroll
      for (int r = 0; r < 8; ++r) {
        int m = m0 + wm * 32 + s * 16 + r + hi8;
        int bb = m >> 11, t = m & (NT - 1);
        dst[(((size_t)(bb * HEADS + h)) * NT + t) * HDIM + d] =
            f2bf(acc[s][nt][r] + bvv);
      }
    }
}

// ------------------------------------------------- flash attention (causal)
// grid: (T/128, H, B)  block: 256 (8 waves, 16 q-rows each)
__global__ __launch_bounds__(256) void attn_fwd(
    const unsigned short* __restrict__ Q, const unsigned short* __restrict__ K,
    const unsigned short* __restrict__ V, unsigned short* __restrict__ Ab) {
  __shared__ __align__(16) unsigned short Vt[64 * 32];        // [d][key]
  __shared__ __align__(16) unsigned short Pl[8 * 16 * 32];    // per-wave P scratch

  const int tid = threadIdx.x;
  const int laneLo = tid & 15;
  const int hi8 = ((tid >> 4) & 1) * 8;
  const int w = tid >> 5;
  const int b = blockIdx.z, h = blockIdx.y;
  const int q0 = blockIdx.x * 128;

  const size_t base = ((size_t)(b * HEADS + h)) * NT * HDIM;
  const unsigned short* Qp = Q + base;
  const unsigned short* Kp = K + base;
  const unsigned short* Vp = V + base;

  // Q fragments: 16 rows x 64 d, kept in registers
  Frag qf[2];
  const int qrow = q0 + w * 16 + laneLo;
#pragma unroll
  for (int t = 0; t < 2; ++t)
    frag_load(qf[t], Qp + (size_t)qrow * HDIM + t * 32 + hi8);

  const v8f vzero = {0.f, 0.f, 0.f, 0.f, 0.f, 0.f, 0.f, 0.f};
  v8f O[4];
#pragma unroll
  for (int i = 0; i < 4; ++i) O[i] = vzero;
  float mrow[8], lrow[8];
#pragma unroll
  for (int r = 0; r < 8; ++r) { mrow[r] = -1.0e30f; lrow[r] = 0.f; }

  const int ntiles = (q0 + 128) >> 5;   // causal: only tiles up to q range
  for (int kt = 0; kt < ntiles; ++kt) {
    const int kbase = kt << 5;
    __syncthreads();
    {  // stage V tile transposed into LDS: Vt[d][key]
      int key = tid >> 3, dc = (tid & 7) * 8;
      uint4 vv = *(const uint4*)(Vp + (size_t)(kbase + key) * HDIM + dc);
      const unsigned short* s = (const unsigned short*)&vv;
#pragma unroll
      for (int j = 0; j < 8; ++j) Vt[(dc + j) * 32 + key] = s[j];
    }
    if (kt + 1 < ntiles)
      __builtin_prefetch(Kp + (size_t)(kbase + 32) * HDIM, 0, 1);
    __syncthreads();

    // S = Q K^T  (two 16x16 tiles covering 32 keys)
    v8f S[2];
#pragma unroll
    for (int st = 0; st < 2; ++st) {
      Frag kf0, kf1;
      int key = kbase + st * 16 + laneLo;
      frag_load(kf0, Kp + (size_t)key * HDIM + 0 + hi8);
      frag_load(kf1, Kp + (size_t)key * HDIM + 32 + hi8);
      v8f z = vzero;
      z = wmma_bf16(qf[0], kf0, z);
      z = wmma_bf16(qf[1], kf1, z);
      S[st] = z;
    }

    // scale + causal mask + row max
    float rmax[8];
#pragma unroll
    for (int r = 0; r < 8; ++r) {
      int q = q0 + w * 16 + r + hi8;
      int kk = kbase + laneLo;
      float s0 = S[0][r] * 0.125f;  if (kk > q)      s0 = -1.0e30f;
      float s1 = S[1][r] * 0.125f;  if (kk + 16 > q) s1 = -1.0e30f;
      S[0][r] = s0; S[1][r] = s1;
      rmax[r] = fmaxf(s0, s1);
    }
#pragma unroll
    for (int r = 0; r < 8; ++r) {
      float v = rmax[r];
#pragma unroll
      for (int off = 1; off < 16; off <<= 1) v = fmaxf(v, __shfl_xor(v, off, 16));
      rmax[r] = v;
    }

    float alpha[8];
#pragma unroll
    for (int r = 0; r < 8; ++r) {
      float nm = fmaxf(mrow[r], rmax[r]);
      alpha[r] = __expf(mrow[r] - nm);
      mrow[r] = nm;
      float p0 = __expf(S[0][r] - nm);
      float p1 = __expf(S[1][r] - nm);
      S[0][r] = p0; S[1][r] = p1;
      float rs = p0 + p1;
#pragma unroll
      for (int off = 1; off < 16; off <<= 1) rs += __shfl_xor(rs, off, 16);
      lrow[r] = lrow[r] * alpha[r] + rs;
    }
#pragma unroll
    for (int nt = 0; nt < 4; ++nt)
#pragma unroll
      for (int r = 0; r < 8; ++r) O[nt][r] *= alpha[r];

    // transpose P through per-wave LDS into A-fragment layout
    unsigned short* myP = Pl + w * (16 * 32);
#pragma unroll
    for (int r = 0; r < 8; ++r) {
      int row = r + hi8;
      myP[row * 32 + laneLo]      = f2bf(S[0][r]);
      myP[row * 32 + 16 + laneLo] = f2bf(S[1][r]);
    }
    Frag pf;
    frag_load(pf, myP + laneLo * 32 + hi8);

    // O += P x V
#pragma unroll
    for (int nt = 0; nt < 4; ++nt) {
      Frag vf;
      frag_load(vf, Vt + (nt * 16 + laneLo) * 32 + hi8);
      O[nt] = wmma_bf16(pf, vf, O[nt]);
    }
  }

  // epilogue: normalize, store bf16 to [B][T][C]
#pragma unroll
  for (int r = 0; r < 8; ++r) {
    int q = q0 + w * 16 + r + hi8;
    float inv = 1.0f / lrow[r];
    size_t rowoff = ((size_t)b * NT + q) * EMB + h * HDIM;
#pragma unroll
    for (int nt = 0; nt < 4; ++nt)
      Ab[rowoff + nt * 16 + laneLo] = f2bf(O[nt][r] * inv);
  }
}

// ----------------------------------------------------- output projection
// grid: (BT/128, EMB/128)  block: 256;  f32 output
__global__ __launch_bounds__(256) void out_proj(
    const unsigned short* __restrict__ Ab, const unsigned short* __restrict__ Wob,
    const float* __restrict__ bo, float* __restrict__ out) {
  __shared__ __align__(16) unsigned short At[128 * 32];
  __shared__ __align__(16) unsigned short Bt[128 * 32];

  const int tid = threadIdx.x;
  const int laneLo = tid & 15;
  const int hi8 = ((tid >> 4) & 1) * 8;
  const int wm = (tid >> 5) & 3;
  const int wn = tid >> 7;
  const int m0 = blockIdx.x * 128;
  const int n0 = blockIdx.y * 128;

  const v8f vzero = {0.f, 0.f, 0.f, 0.f, 0.f, 0.f, 0.f, 0.f};
  v8f acc[2][4];
#pragma unroll
  for (int s = 0; s < 2; ++s)
#pragma unroll
    for (int i = 0; i < 4; ++i) acc[s][i] = vzero;

  for (int k0 = 0; k0 < EMB; k0 += 32) {
    __syncthreads();
#pragma unroll
    for (int i = 0; i < 2; ++i) {
      int c = tid + i * 256;
      int row = c >> 2;
      int kc = (c & 3) * 8;
      async_copy_b128(Ab + (size_t)(m0 + row) * EMB + k0 + kc,
                      At + row * 32 + kc);
    }
#pragma unroll
    for (int i = 0; i < 2; ++i) {
      int c = tid + i * 256;
      int k = c >> 4;
      int nc = (c & 15) * 8;
      uint4 vv = *(const uint4*)(Wob + (size_t)(k0 + k) * EMB + n0 + nc);
      const unsigned short* s = (const unsigned short*)&vv;
#pragma unroll
      for (int j = 0; j < 8; ++j) Bt[(nc + j) * 32 + k] = s[j];
    }
    wait_async0();
    __syncthreads();

    Frag a[2];
    frag_load(a[0], At + (wm * 32 + laneLo) * 32 + hi8);
    frag_load(a[1], At + (wm * 32 + 16 + laneLo) * 32 + hi8);
#pragma unroll
    for (int nt = 0; nt < 4; ++nt) {
      Frag b;
      frag_load(b, Bt + (wn * 64 + nt * 16 + laneLo) * 32 + hi8);
      acc[0][nt] = wmma_bf16(a[0], b, acc[0][nt]);
      acc[1][nt] = wmma_bf16(a[1], b, acc[1][nt]);
    }
  }

#pragma unroll
  for (int s = 0; s < 2; ++s)
#pragma unroll
    for (int nt = 0; nt < 4; ++nt) {
      int n = n0 + wn * 64 + nt * 16 + laneLo;
      float bvv = bo[n];
#pragma unroll
      for (int r = 0; r < 8; ++r) {
        int m = m0 + wm * 32 + s * 16 + r + hi8;
        out[(size_t)m * EMB + n] = acc[s][nt][r] + bvv;
      }
    }
}

// ---------------------------------------------------------------- launch
extern "C" void kernel_launch(void* const* d_in, const int* in_sizes, int n_in,
                              void* d_out, int out_size, void* d_ws, size_t ws_size,
                              hipStream_t stream) {
  const float* x  = (const float*)d_in[0];
  const float* Wq = (const float*)d_in[1];
  const float* bq = (const float*)d_in[2];
  const float* Wk = (const float*)d_in[3];
  const float* bk = (const float*)d_in[4];
  const float* Wv = (const float*)d_in[5];
  const float* bv = (const float*)d_in[6];
  const float* Wo = (const float*)d_in[7];
  const float* bo = (const float*)d_in[8];
  float* out = (float*)d_out;

  char* ws = (char*)d_ws;
  const size_t MB = 1024 * 1024;
  unsigned short* xb  = (unsigned short*)(ws + 0);        //  8 MB
  unsigned short* Wqb = (unsigned short*)(ws + 8 * MB);   //  2 MB
  unsigned short* Wkb = (unsigned short*)(ws + 10 * MB);
  unsigned short* Wvb = (unsigned short*)(ws + 12 * MB);
  unsigned short* Wob = (unsigned short*)(ws + 14 * MB);
  unsigned short* Qb  = (unsigned short*)(ws + 16 * MB);  //  8 MB each
  unsigned short* Kb  = (unsigned short*)(ws + 24 * MB);
  unsigned short* Vb  = (unsigned short*)(ws + 32 * MB);
  unsigned short* Ab  = (unsigned short*)(ws + 40 * MB);  // -> 48 MB total

  const int NX = BT * EMB;         // 4,194,304
  const int NW = EMB * EMB;        // 1,048,576
  cvt_bf16<<<(NX + 255) / 256, 256, 0, stream>>>(x,  xb,  NX);
  cvt_bf16<<<(NW + 255) / 256, 256, 0, stream>>>(Wq, Wqb, NW);
  cvt_bf16<<<(NW + 255) / 256, 256, 0, stream>>>(Wk, Wkb, NW);
  cvt_bf16<<<(NW + 255) / 256, 256, 0, stream>>>(Wv, Wvb, NW);
  cvt_bf16<<<(NW + 255) / 256, 256, 0, stream>>>(Wo, Wob, NW);

  qkv_gemm<<<dim3(BT / 128, EMB / 128, 3), 256, 0, stream>>>(
      xb, Wqb, Wkb, Wvb, bq, bk, bv, Qb, Kb, Vb);

  attn_fwd<<<dim3(NT / 128, HEADS, NB), 256, 0, stream>>>(Qb, Kb, Vb, Ab);

  out_proj<<<dim3(BT / 128, EMB / 128), 256, 0, stream>>>(Ab, Wob, bo, out);
}
